// MultiHeadAttention_60516089201062
// MI455X (gfx1250) — matrix operand order repaired and checked by reading, not compiled, not run on hardware
//
#include <hip/hip_runtime.h>
#include <hip/hip_bf16.h>

typedef __attribute__((ext_vector_type(16))) _Float16 v16h;
typedef __attribute__((ext_vector_type(8)))  _Float16 v8h;
typedef __attribute__((ext_vector_type(8)))  float    v8f;
typedef __attribute__((ext_vector_type(4)))  float    v4f;

#define D_MODEL   768
#define NUM_HEADS 12
#define D_K       64
#define SEQ       2048
#define BATCH     2
#define M_TOT     (BATCH * SEQ)   // 4096 tokens

// ---------------------------------------------------------------------------
// WMMA helpers (CDNA5 16x16x32 f16 -> f32, wave32 layouts per ISA 7.12.2)
// ---------------------------------------------------------------------------
__device__ __forceinline__ v8f wmma_f16(v16h a, v16h b, v8f c) {
  // D = A*B + C ; args: (neg_a, A, neg_b, B, c_mod, C, reuse_a, reuse_b)
  return __builtin_amdgcn_wmma_f32_16x16x32_f16(false, a, false, b, (short)0, c,
                                                false, false);
}

// A fragment: 16(M) x 32(K) f16, row-major source [.. x ld].
// lane L: row = L&15 ; element i <-> K = (i<8 ? i : 16+i-8) + (L>>4)*8
__device__ __forceinline__ v16h load_a_h(const _Float16* p, int ld, int lane) {
  int row = lane & 15, half = lane >> 4;
  const _Float16* r = p + row * ld + half * 8;
  v8h lo = *(const v8h*)(r);
  v8h hi = *(const v8h*)(r + 16);
  v16h a;
#pragma unroll
  for (int i = 0; i < 8; ++i) { a[i] = lo[i]; a[8 + i] = hi[i]; }
  return a;
}

// B fragment: 32(K) x 16(N) where B = X^T and X is row-major [16(N) x ld(K)].
// lane L: n = L&15 ; element i <-> K = (i<8 ? i : 16+i-8) + (L>>4)*8
__device__ __forceinline__ v16h load_bt_h(const _Float16* p, int ld, int lane) {
  int n = lane & 15, half = lane >> 4;
  const _Float16* r = p + n * ld + half * 8;
  v8h lo = *(const v8h*)(r);
  v8h hi = *(const v8h*)(r + 16);
  v16h b;
#pragma unroll
  for (int i = 0; i < 8; ++i) { b[i] = lo[i]; b[8 + i] = hi[i]; }
  return b;
}
// C/D fragment v8f: element r <-> D[row = r + (L>>4)*8, col = L&15]

// ---------------------------------------------------------------------------
// Pass 0a: activation convert f32 -> f16 (4 elements / thread)
// ---------------------------------------------------------------------------
__global__ __launch_bounds__(256) void cvt_h_kernel(const float* __restrict__ X,
                                                    _Float16* __restrict__ Y) {
  int i = (blockIdx.x * 256 + threadIdx.x) * 4;
  v4f x = *(const v4f*)(X + i);
  _Float16 y0 = (_Float16)x[0], y1 = (_Float16)x[1];
  _Float16 y2 = (_Float16)x[2], y3 = (_Float16)x[3];
  __attribute__((ext_vector_type(4))) _Float16 y = {y0, y1, y2, y3};
  *(__attribute__((ext_vector_type(4))) _Float16*)(Y + i) = y;
}

// ---------------------------------------------------------------------------
// Pass 0b: weight pack  Wt[n][k] = (f16) W[k][n]
// ---------------------------------------------------------------------------
__global__ __launch_bounds__(256) void pack_w_kernel(const float* __restrict__ W,
                                                     _Float16* __restrict__ Wt) {
  int idx = blockIdx.x * 256 + threadIdx.x;
  if (idx < D_MODEL * D_MODEL) {
    int k = idx / D_MODEL, n = idx % D_MODEL;       // coalesced read over n
    Wt[n * D_MODEL + k] = (_Float16)W[idx];
  }
}

// ---------------------------------------------------------------------------
// Pass 1: fused QKV projections.  z=0:Q  z=1:K  z=2:V(transposed)
// Each wave computes a 64x64 tile; block = 4 waves = 128x128 tile.
// ---------------------------------------------------------------------------
__global__ __launch_bounds__(128) void qkv_proj_kernel(
    const _Float16* __restrict__ Xq, const _Float16* __restrict__ Xk,
    const _Float16* __restrict__ Xv,
    const _Float16* __restrict__ Wtq, const _Float16* __restrict__ Wtk,
    const _Float16* __restrict__ Wtv,
    const float* __restrict__ bq, const float* __restrict__ bk,
    const float* __restrict__ bv,
    _Float16* __restrict__ Qh, _Float16* __restrict__ Kh,
    _Float16* __restrict__ Vt) {
  int z = blockIdx.z;
  const _Float16* X    = (z == 0) ? Xq  : (z == 1) ? Xk  : Xv;
  const _Float16* Wt   = (z == 0) ? Wtq : (z == 1) ? Wtk : Wtv;
  const float*    bias = (z == 0) ? bq  : (z == 1) ? bk  : bv;

  int lane = threadIdx.x & 31;
  int wave = threadIdx.x >> 5;
  int wm = wave >> 1, wn = wave & 1;
  int m0 = blockIdx.x * 128 + wm * 64;
  int n0 = blockIdx.y * 128 + wn * 64;

  v8f acc[4][4] = {};
#pragma unroll 1
  for (int k0 = 0; k0 < D_MODEL; k0 += 32) {
    v16h a[4], b[4];
#pragma unroll
    for (int mi = 0; mi < 4; ++mi)
      a[mi] = load_a_h(X + (m0 + mi * 16) * D_MODEL + k0, D_MODEL, lane);
#pragma unroll
    for (int ni = 0; ni < 4; ++ni)
      b[ni] = load_bt_h(Wt + (n0 + ni * 16) * D_MODEL + k0, D_MODEL, lane);
#pragma unroll
    for (int mi = 0; mi < 4; ++mi)
#pragma unroll
      for (int ni = 0; ni < 4; ++ni)
        acc[mi][ni] = wmma_f16(a[mi], b[ni], acc[mi][ni]);
  }

  int colL = lane & 15, half = lane >> 4;
#pragma unroll
  for (int mi = 0; mi < 4; ++mi) {
#pragma unroll
    for (int ni = 0; ni < 4; ++ni) {
      int n = n0 + ni * 16 + colL;
      float bval = bias[n];
      int h = n >> 6, d = n & 63;                    // feature -> (head, dk)
#pragma unroll
      for (int r = 0; r < 8; ++r) {
        int m = m0 + mi * 16 + half * 8 + r;         // token row
        int bb = m >> 11, s = m & (SEQ - 1);
        float val = acc[mi][ni][r] + bval;
        if (z == 0) {
          Qh[((bb * NUM_HEADS + h) * SEQ + s) * D_K + d] = (_Float16)val;
        } else if (z == 1) {
          Kh[((bb * NUM_HEADS + h) * SEQ + s) * D_K + d] = (_Float16)val;
        } else {
          Vt[((bb * NUM_HEADS + h) * D_K + d) * SEQ + s] = (_Float16)val;
        }
      }
    }
  }
}

// ---------------------------------------------------------------------------
// Pass 2: flash attention.  One wave per TWO 16-query tiles (32 queries),
// 32 keys per step; K/V fragments reused across both query tiles.
// ---------------------------------------------------------------------------
__global__ __launch_bounds__(128) void attn_kernel(
    const _Float16* __restrict__ Qh, const _Float16* __restrict__ Kh,
    const _Float16* __restrict__ Vt, _Float16* __restrict__ O) {
  __shared__ _Float16 pbuf[4][32 * 32];   // per-wave P tile (32 q x 32 keys)

  int lane = threadIdx.x & 31;
  int wave = threadIdx.x >> 5;
  int task  = blockIdx.x * 4 + wave;      // 0 .. B*H*(S/32)-1
  int qpair = task & (SEQ / 32 - 1);      // 0..63
  int bh    = task >> 6;                  // 0..23

  const _Float16* Qp = Qh + (bh * SEQ + qpair * 32) * D_K;
  const _Float16* Kp = Kh + bh * SEQ * D_K;
  const _Float16* Vp = Vt + bh * D_K * SEQ;

  int colL = lane & 15, half = lane >> 4;
  _Float16* myp = pbuf[wave];

  // Q fragments: 2 query tiles x 2 dk-halves, fold in 1/sqrt(dk)=0.125 exactly
  v16h qa[2][2];
#pragma unroll
  for (int t = 0; t < 2; ++t)
#pragma unroll
    for (int kk = 0; kk < 2; ++kk) {
      qa[t][kk] = load_a_h(Qp + t * 16 * D_K + kk * 32, D_K, lane);
#pragma unroll
      for (int i = 0; i < 16; ++i)
        qa[t][kk][i] = qa[t][kk][i] * (_Float16)0.125f;
    }

  float m_[2][8], l_[2][8];
#pragma unroll
  for (int t = 0; t < 2; ++t)
#pragma unroll
    for (int r = 0; r < 8; ++r) { m_[t][r] = -__builtin_inff(); l_[t][r] = 0.0f; }
  v8f acc[2][4] = {};

#pragma unroll 1
  for (int s0 = 0; s0 < SEQ; s0 += 32) {
    // ---- K fragments for this 32-key step (shared by both query tiles) ----
    v16h kb[2][2];
#pragma unroll
    for (int j = 0; j < 2; ++j) {
      const _Float16* kp = Kp + (s0 + j * 16) * D_K;
      kb[j][0] = load_bt_h(kp, D_K, lane);        // dk  0..31
      kb[j][1] = load_bt_h(kp + 32, D_K, lane);   // dk 32..63
    }
    // ---- scores: 2 q-tiles x 2 key-tiles, contraction over dk=64 ----
    v8f sc[2][2] = {};
#pragma unroll
    for (int t = 0; t < 2; ++t)
#pragma unroll
      for (int j = 0; j < 2; ++j) {
        sc[t][j] = wmma_f16(qa[t][0], kb[j][0], sc[t][j]);
        sc[t][j] = wmma_f16(qa[t][1], kb[j][1], sc[t][j]);
      }
    // ---- online softmax over the 32 keys, per query tile ----
#pragma unroll
    for (int t = 0; t < 2; ++t) {
      float alpha[8];
#pragma unroll
      for (int r = 0; r < 8; ++r) {
        float rm = fmaxf(sc[t][0][r], sc[t][1][r]);
#pragma unroll
        for (int off = 8; off >= 1; off >>= 1)
          rm = fmaxf(rm, __shfl_xor(rm, off, 16));
        float mn = fmaxf(m_[t][r], rm);
        alpha[r] = __expf(m_[t][r] - mn);
        float p0 = __expf(sc[t][0][r] - mn);
        float p1 = __expf(sc[t][1][r] - mn);
        sc[t][0][r] = p0; sc[t][1][r] = p1;
        float rs = p0 + p1;
#pragma unroll
        for (int off = 8; off >= 1; off >>= 1)
          rs += __shfl_xor(rs, off, 16);
        l_[t][r] = l_[t][r] * alpha[r] + rs;
        m_[t][r] = mn;
      }
#pragma unroll
      for (int ni = 0; ni < 4; ++ni)
#pragma unroll
        for (int r = 0; r < 8; ++r) acc[t][ni][r] *= alpha[r];
      // P: C-layout -> LDS (rows t*16 .. t*16+15)
#pragma unroll
      for (int r = 0; r < 8; ++r) {
        myp[(t * 16 + r + half * 8) * 32 + colL]      = (_Float16)sc[t][0][r];
        myp[(t * 16 + r + half * 8) * 32 + 16 + colL] = (_Float16)sc[t][1][r];
      }
    }
    __syncthreads();
    // ---- P as A-fragments (transpose across lanes via LDS) ----
    v16h pa[2];
#pragma unroll
    for (int t = 0; t < 2; ++t) {
      const _Float16* pr = myp + (t * 16 + colL) * 32 + half * 8;
      v8h lo = *(const v8h*)(pr);
      v8h hi = *(const v8h*)(pr + 16);
#pragma unroll
      for (int i = 0; i < 8; ++i) { pa[t][i] = lo[i]; pa[t][8 + i] = hi[i]; }
    }
    // ---- acc += P (32q x 32keys) @ V (32keys x 64dk) ----
#pragma unroll
    for (int ni = 0; ni < 4; ++ni) {
      v16h vb = load_bt_h(Vp + (ni * 16) * SEQ + s0, SEQ, lane);
      acc[0][ni] = wmma_f16(pa[0], vb, acc[0][ni]);
      acc[1][ni] = wmma_f16(pa[1], vb, acc[1][ni]);
    }
    __syncthreads();
  }

  // ---- epilogue: normalize and store context [B,H,S,dk] f16 ----
#pragma unroll
  for (int t = 0; t < 2; ++t)
#pragma unroll
    for (int ni = 0; ni < 4; ++ni)
#pragma unroll
      for (int r = 0; r < 8; ++r) {
        int q = qpair * 32 + t * 16 + half * 8 + r;
        int d = ni * 16 + colL;
        O[(bh * SEQ + q) * D_K + d] = (_Float16)(acc[t][ni][r] / l_[t][r]);
      }
}

// ---------------------------------------------------------------------------
// Pass 3: output projection  out[m][n] = ctx[m][:] @ Wo[:][n] + bo[n]  (f32)
// ctx feature k -> head h=k/64, d=k%64 in O[B,H,S,64]
// ---------------------------------------------------------------------------
__global__ __launch_bounds__(128) void out_proj_kernel(
    const _Float16* __restrict__ O, const _Float16* __restrict__ Wto,
    const float* __restrict__ bo, float* __restrict__ out) {
  int lane = threadIdx.x & 31;
  int wave = threadIdx.x >> 5;
  int wm = wave >> 1, wn = wave & 1;
  int m0 = blockIdx.x * 128 + wm * 64;
  int n0 = blockIdx.y * 128 + wn * 64;
  int colL = lane & 15, half = lane >> 4;

  v8f acc[4][4] = {};
#pragma unroll 1
  for (int k0 = 0; k0 < D_MODEL; k0 += 32) {
    int h = k0 >> 6, dbase = k0 & 63;   // a 32-wide K-step never straddles a head
    v16h a[4], b[4];
#pragma unroll
    for (int mi = 0; mi < 4; ++mi) {
      int m = m0 + mi * 16 + colL;      // A row = lane&15
      int bb = m >> 11, s = m & (SEQ - 1);
      const _Float16* base =
          O + ((bb * NUM_HEADS + h) * SEQ + s) * D_K + dbase;
      v8h lo = *(const v8h*)(base + half * 8);
      v8h hi = *(const v8h*)(base + 16 + half * 8);
#pragma unroll
      for (int i = 0; i < 8; ++i) { a[mi][i] = lo[i]; a[mi][8 + i] = hi[i]; }
    }
#pragma unroll
    for (int ni = 0; ni < 4; ++ni)
      b[ni] = load_bt_h(Wto + (n0 + ni * 16) * D_MODEL + k0, D_MODEL, lane);
#pragma unroll
    for (int mi = 0; mi < 4; ++mi)
#pragma unroll
      for (int ni = 0; ni < 4; ++ni)
        acc[mi][ni] = wmma_f16(a[mi], b[ni], acc[mi][ni]);
  }

#pragma unroll
  for (int mi = 0; mi < 4; ++mi) {
#pragma unroll
    for (int ni = 0; ni < 4; ++ni) {
      int n = n0 + ni * 16 + colL;
      float bval = bo[n];
#pragma unroll
      for (int r = 0; r < 8; ++r) {
        int m = m0 + mi * 16 + half * 8 + r;
        out[m * D_MODEL + n] = acc[mi][ni][r] + bval;
      }
    }
  }
}

// ---------------------------------------------------------------------------
extern "C" void kernel_launch(void* const* d_in, const int* in_sizes, int n_in,
                              void* d_out, int out_size, void* d_ws,
                              size_t ws_size, hipStream_t stream) {
  const float* query = (const float*)d_in[0];
  const float* key   = (const float*)d_in[1];
  const float* value = (const float*)d_in[2];
  const float* Wq = (const float*)d_in[3];
  const float* bq = (const float*)d_in[4];
  const float* Wk = (const float*)d_in[5];
  const float* bk = (const float*)d_in[6];
  const float* Wv = (const float*)d_in[7];
  const float* bv = (const float*)d_in[8];
  const float* Wo = (const float*)d_in[9];
  const float* bo = (const float*)d_in[10];
  float* out = (float*)d_out;

  // workspace carve (all 256B aligned)
  size_t off = 0;
  auto carve = [&](size_t bytes) {
    void* p = (char*)d_ws + off;
    off += (bytes + 255) & ~(size_t)255;
    return p;
  };
  const size_t wbytes = (size_t)D_MODEL * D_MODEL * sizeof(_Float16);
  const size_t xbytes = (size_t)M_TOT * D_MODEL * sizeof(_Float16);
  const size_t hbytes = (size_t)BATCH * NUM_HEADS * SEQ * D_K * sizeof(_Float16);
  _Float16* Wtq = (_Float16*)carve(wbytes);
  _Float16* Wtk = (_Float16*)carve(wbytes);
  _Float16* Wtv = (_Float16*)carve(wbytes);
  _Float16* Wto = (_Float16*)carve(wbytes);
  _Float16* Xhq = (_Float16*)carve(xbytes);
  _Float16* Xhk = (_Float16*)carve(xbytes);
  _Float16* Xhv = (_Float16*)carve(xbytes);
  _Float16* Qh  = (_Float16*)carve(hbytes);
  _Float16* Kh  = (_Float16*)carve(hbytes);
  _Float16* Vt  = (_Float16*)carve(hbytes);   // [B,H,64,S]
  _Float16* Oc  = (_Float16*)carve(hbytes);   // context [B,H,S,64]
  (void)ws_size; (void)n_in; (void)in_sizes; (void)out_size;

  // Pass 0: convert activations to f16; pack weights (f32 [k][n] -> f16 [n][k])
  int cvt_grid = (M_TOT * D_MODEL) / (256 * 4);
  cvt_h_kernel<<<cvt_grid, 256, 0, stream>>>(query, Xhq);
  cvt_h_kernel<<<cvt_grid, 256, 0, stream>>>(key, Xhk);
  cvt_h_kernel<<<cvt_grid, 256, 0, stream>>>(value, Xhv);
  int pack_grid = (D_MODEL * D_MODEL + 255) / 256;
  pack_w_kernel<<<pack_grid, 256, 0, stream>>>(Wq, Wtq);
  pack_w_kernel<<<pack_grid, 256, 0, stream>>>(Wk, Wtk);
  pack_w_kernel<<<pack_grid, 256, 0, stream>>>(Wv, Wtv);
  pack_w_kernel<<<pack_grid, 256, 0, stream>>>(Wo, Wto);

  // Pass 1: QKV projections
  qkv_proj_kernel<<<dim3(M_TOT / 128, D_MODEL / 128, 3), 128, 0, stream>>>(
      Xhq, Xhk, Xhv, Wtq, Wtk, Wtv, bq, bk, bv, Qh, Kh, Vt);

  // Pass 2: flash attention (B*H*(S/32) = 1536 wave-tasks, 4 waves/block)
  attn_kernel<<<dim3(BATCH * NUM_HEADS * (SEQ / 32) / 4), 128, 0, stream>>>(
      Qh, Kh, Vt, Oc);

  // Pass 3: output projection -> f32
  out_proj_kernel<<<dim3(M_TOT / 128, D_MODEL / 128), 128, 0, stream>>>(
      Oc, Wto, bo, out);
}
